// MultiHeadAttention_231928234515
// MI455X (gfx1250) — matrix operand order repaired and checked by reading, not compiled, not run on hardware
//
#include <hip/hip_runtime.h>

typedef __attribute__((ext_vector_type(16))) _Float16 v16h;
typedef __attribute__((ext_vector_type(8)))  _Float16 v8h;
typedef __attribute__((ext_vector_type(8)))  float    v8f;

#define D_MODEL 1024
#define NHEAD   16
#define HD      64
#define SEQ     2048
#define BATCH   2

#define SHUF16 0,1,2,3,4,5,6,7,8,9,10,11,12,13,14,15

static __device__ __forceinline__ v8f wmma_f16(v16h a, v16h b, v8f c) {
  // D = A(16x32 f16) * B(32x16 f16) + C(16x16 f32)
  return __builtin_amdgcn_wmma_f32_16x16x32_f16(
      /*neg_a=*/false, a, /*neg_b=*/false, b,
      /*c_mod=*/(short)0, c, /*reuse_a=*/false, /*reuse_b=*/false);
}

// ---------------------------------------------------------------------------
// fp32 -> f16 pre-pass (bandwidth-bound; removes v_cvt from the GEMM loops).
// ---------------------------------------------------------------------------
__global__ __launch_bounds__(256) void cvt_f16_kernel(
    const float* __restrict__ src, _Float16* __restrict__ dst)
{
  const size_t i = ((size_t)blockIdx.x * 256 + threadIdx.x) * 8;
  float4 a = *(const float4*)(src + i);
  float4 b = *(const float4*)(src + i + 4);
  v8h o;
  o[0] = (_Float16)a.x; o[1] = (_Float16)a.y; o[2] = (_Float16)a.z; o[3] = (_Float16)a.w;
  o[4] = (_Float16)b.x; o[5] = (_Float16)b.y; o[6] = (_Float16)b.z; o[7] = (_Float16)b.w;
  *(v8h*)(dst + i) = o;
}

// ---------------------------------------------------------------------------
// Projection: out = (X @ W^T + bias) * scale, f16 in / f16 out.
// One wave computes a 64x16 strip: 4 WMMAs per K-step sharing one B-fragment.
//   VT==0: out[bh][s][d]   (Q, K)        VT==1: out[bh][d][s]  (V transposed)
// ---------------------------------------------------------------------------
template <int VT>
__global__ __launch_bounds__(32) void proj_kernel(
    const _Float16* __restrict__ Xh, const _Float16* __restrict__ Wh,
    const float* __restrict__ bias, _Float16* __restrict__ out, float scale)
{
  const int lane = threadIdx.x;
  const int lm = lane & 15;          // A row / B col / C col this lane owns
  const int kh = lane >> 4;          // lane half selects K sub-range
  const int mt = blockIdx.x;         // 64-row strip over B*SEQ   (64)
  const int nt = blockIdx.y;         // 16-col tile over D_MODEL  (64)

  const _Float16* __restrict__ xr[4];
#pragma unroll
  for (int st = 0; st < 4; ++st)
    xr[st] = Xh + (size_t)(mt * 64 + st * 16 + lm) * D_MODEL;
  const _Float16* __restrict__ wr = Wh + (size_t)(nt * 16 + lm) * D_MODEL;

  v8f acc[4] = {};
  for (int k0 = 0; k0 < D_MODEL; k0 += 32) {
    // B-fragment: lane group kh holds K kh*16..kh*16+15 of column lm
    v16h bf = *(const v16h*)(wr + k0 + kh * 16);
#pragma unroll
    for (int st = 0; st < 4; ++st) {
      // A-fragment: lane<16 holds K 0..15, lane>=16 holds 16..31
      v8h lo = *(const v8h*)(xr[st] + k0 + kh * 16);
      v8h hi = *(const v8h*)(xr[st] + k0 + kh * 16 + 8);
      v16h a = __builtin_shufflevector(lo, hi, SHUF16);
      acc[st] = wmma_f16(a, bf, acc[st]);
    }
  }

  const int ncol = nt * 16 + lm;     // C layout: lane owns one column
  const float bv = bias[ncol];
  const int h = ncol >> 6, d = ncol & (HD - 1);

#pragma unroll
  for (int st = 0; st < 4; ++st) {
    if (VT) {
      // rows are contiguous along SEQ in the transposed layout -> 16B store
      v8h o8;
#pragma unroll
      for (int r = 0; r < 8; ++r) o8[r] = (_Float16)((acc[st][r] + bv) * scale);
      const int m0 = mt * 64 + st * 16 + kh * 8;
      const int bidx = m0 >> 11, s = m0 & (SEQ - 1);
      *(v8h*)(out + (((size_t)bidx * NHEAD + h) * HD + d) * SEQ + s) = o8;
    } else {
#pragma unroll
      for (int r = 0; r < 8; ++r) {
        const int mrow = mt * 64 + st * 16 + r + kh * 8;
        const int bidx = mrow >> 11, s = mrow & (SEQ - 1);
        out[(((size_t)bidx * NHEAD + h) * SEQ + s) * HD + d] =
            (_Float16)((acc[st][r] + bv) * scale);
      }
    }
  }
}

// ---------------------------------------------------------------------------
// Flash attention, one wave per (batch*head, 16-query tile).
// Q was pre-scaled by log2(e)/sqrt(HD), so softmax uses raw v_exp_f32 (2^x).
// Row sums are kept as per-lane partials and reduced once after the K loop.
// ---------------------------------------------------------------------------
__global__ __launch_bounds__(32) void attn_kernel(
    const _Float16* __restrict__ Qw, const _Float16* __restrict__ Kw,
    const _Float16* __restrict__ Vt, const int* __restrict__ mask,
    float* __restrict__ out)
{
  __shared__ _Float16 pbuf[16 * 32];           // P tile staging (C-layout -> A-layout)
  const int lane = threadIdx.x;
  const int lm = lane & 15;
  const int kh = lane >> 4;
  const int bh = blockIdx.x;                   // 0..B*H-1
  const int qt = blockIdx.y;                   // 0..SEQ/16-1
  const int b  = bh >> 4;

  // Q A-fragments for the two K-steps of HD=64 (already scaled in proj)
  v16h qf[2];
  {
    const _Float16* qrow = Qw + ((size_t)bh * SEQ + qt * 16 + lm) * HD;
#pragma unroll
    for (int t = 0; t < 2; ++t) {
      v8h lo = *(const v8h*)(qrow + t * 32 + kh * 16);
      v8h hi = *(const v8h*)(qrow + t * 32 + kh * 16 + 8);
      qf[t] = __builtin_shufflevector(lo, hi, SHUF16);
    }
  }

  v8f o[4] = {};                               // 16x64 f32 output accumulator
  float mst[8], lst[8];                        // running row max / per-lane partial sum
#pragma unroll
  for (int r = 0; r < 8; ++r) { mst[r] = -3.0e38f; lst[r] = 0.0f; }

  const int* mbase = mask + ((size_t)b * SEQ + qt * 16 + kh * 8) * SEQ;

  for (int kt = 0; kt < SEQ; kt += 32) {
    if (kt + 32 < SEQ) {                       // uniform branch: prefetch next chunk
      __builtin_prefetch(Kw + ((size_t)bh * SEQ + kt + 32 + lm) * HD, 0, 1);
      __builtin_prefetch(Vt + ((size_t)bh * HD + lm) * SEQ + kt + 32, 0, 1);
    }

    // ---- scores (log2 domain): two 16-key sub-tiles, 2 k-steps each ------
    v8f sc2[2];
#pragma unroll
    for (int j = 0; j < 2; ++j) {
      const int key = kt + j * 16 + lm;        // B-fragment column = key index
      const _Float16* krow = Kw + ((size_t)bh * SEQ + key) * HD;
      v8f s = {};
#pragma unroll
      for (int ks = 0; ks < 2; ++ks) {
        v16h bf = *(const v16h*)(krow + ks * 32 + kh * 16);   // 32B-aligned
        s = wmma_f16(qf[ks], bf, s);
      }
      const int* mp = mbase + key;
#pragma unroll
      for (int r = 0; r < 8; ++r)              // branch-free mask (EXEC stays all-1s)
        s[r] = (mp[(size_t)r * SEQ] == 0) ? -1.5e9f : s[r];
      sc2[j] = s;
    }

    // ---- online softmax: only the row max needs cross-lane work ----------
#pragma unroll
    for (int r = 0; r < 8; ++r) {
      float v = fmaxf(sc2[0][r], sc2[1][r]);
      v = fmaxf(v, __shfl_xor(v, 1, 32));      // reduce within 16-lane half
      v = fmaxf(v, __shfl_xor(v, 2, 32));
      v = fmaxf(v, __shfl_xor(v, 4, 32));
      v = fmaxf(v, __shfl_xor(v, 8, 32));
      const float mnew  = fmaxf(mst[r], v);
      const float alpha = __builtin_amdgcn_exp2f(mst[r] - mnew);
      const float p0 = __builtin_amdgcn_exp2f(sc2[0][r] - mnew);
      const float p1 = __builtin_amdgcn_exp2f(sc2[1][r] - mnew);
      lst[r] = fmaf(lst[r], alpha, p0 + p1);   // per-lane partial row sum
      mst[r] = mnew;
#pragma unroll
      for (int t = 0; t < 4; ++t) o[t][r] *= alpha;
      const int prow = r + kh * 8;             // C-layout row this lane holds
      pbuf[prow * 32 + lm]      = (_Float16)p0;
      pbuf[prow * 32 + 16 + lm] = (_Float16)p1;
    }
    __syncthreads();

    // ---- P tile as 16x32 A-fragment (two 16B LDS reads) ------------------
    v8h plo = *(const v8h*)(pbuf + lm * 32 + kh * 16);
    v8h phi = *(const v8h*)(pbuf + lm * 32 + kh * 16 + 8);
    v16h pf = __builtin_shufflevector(plo, phi, SHUF16);
    __syncthreads();

    // ---- O += P @ V (V stored transposed: contiguous B-fragments) --------
#pragma unroll
    for (int t = 0; t < 4; ++t) {
      const _Float16* vp = Vt + ((size_t)bh * HD + t * 16 + lm) * SEQ + kt + kh * 16;
      v16h vf = *(const v16h*)vp;              // 32B-aligned
      o[t] = wmma_f16(pf, vf, o[t]);
    }
  }

  // ---- epilogue: finish row-sum reduction, divide, write [B,S,D] fp32 -----
  const int h = bh & (NHEAD - 1);
#pragma unroll
  for (int r = 0; r < 8; ++r) {
    float l = lst[r];
    l += __shfl_xor(l, 1, 32);
    l += __shfl_xor(l, 2, 32);
    l += __shfl_xor(l, 4, 32);
    l += __shfl_xor(l, 8, 32);
    const float inv = 1.0f / l;
    const int srow = qt * 16 + r + kh * 8;
    float* op = out + ((size_t)b * SEQ + srow) * D_MODEL + h * HD + lm;
#pragma unroll
    for (int t = 0; t < 4; ++t)
      op[t * 16] = o[t][r] * inv;
  }
}

// ---------------------------------------------------------------------------
extern "C" void kernel_launch(void* const* d_in, const int* in_sizes, int n_in,
                              void* d_out, int out_size, void* d_ws, size_t ws_size,
                              hipStream_t stream) {
  const float* query = (const float*)d_in[0];
  const float* key   = (const float*)d_in[1];
  const float* value = (const float*)d_in[2];
  const int*   mask  = (const int*)d_in[3];
  const float* Wq = (const float*)d_in[4];
  const float* bq = (const float*)d_in[5];
  const float* Wk = (const float*)d_in[6];
  const float* bk = (const float*)d_in[7];
  const float* Wv = (const float*)d_in[8];
  const float* bv = (const float*)d_in[9];
  float* out = (float*)d_out;

  const size_t n_xe = (size_t)BATCH * SEQ * D_MODEL;        // 4M elements
  const size_t n_we = (size_t)D_MODEL * D_MODEL;            // 1M elements

  _Float16* p = (_Float16*)d_ws;
  _Float16 *Xh[3], *Wh[3];
  for (int i = 0; i < 3; ++i) { Xh[i] = p; p += n_xe; }
  for (int i = 0; i < 3; ++i) { Wh[i] = p; p += n_we; }
  _Float16* Qw = p;            p += n_xe;
  _Float16* Kw = p;            p += n_xe;
  _Float16* Vt = p;            /* total ~56.6 MB of workspace */

  // fp32 -> f16 pre-pass
  cvt_f16_kernel<<<n_xe / (256 * 8), 256, 0, stream>>>(query, Xh[0]);
  cvt_f16_kernel<<<n_xe / (256 * 8), 256, 0, stream>>>(key,   Xh[1]);
  cvt_f16_kernel<<<n_xe / (256 * 8), 256, 0, stream>>>(value, Xh[2]);
  cvt_f16_kernel<<<n_we / (256 * 8), 256, 0, stream>>>(Wq, Wh[0]);
  cvt_f16_kernel<<<n_we / (256 * 8), 256, 0, stream>>>(Wk, Wh[1]);
  cvt_f16_kernel<<<n_we / (256 * 8), 256, 0, stream>>>(Wv, Wh[2]);

  // Q carries softmax scaling: log2(e)/sqrt(HD)
  const float qscale = 0.125f * 1.4426950408889634f;
  dim3 pg(BATCH * SEQ / 64, D_MODEL / 16);                  // 64 x 64 waves
  proj_kernel<0><<<pg, 32, 0, stream>>>(Xh[0], Wh[0], bq, Qw, qscale);
  proj_kernel<0><<<pg, 32, 0, stream>>>(Xh[1], Wh[1], bk, Kw, 1.0f);
  proj_kernel<1><<<pg, 32, 0, stream>>>(Xh[2], Wh[2], bv, Vt, 1.0f);

  dim3 ag(BATCH * NHEAD, SEQ / 16);                         // 32 x 128 waves
  attn_kernel<<<ag, 32, 0, stream>>>(Qw, Kw, Vt, mask, out);
}
